// StandardTopKMoE_7378753815191
// MI455X (gfx1250) — matrix operand order repaired and checked by reading, not compiled, not run on hardware
//
#include <hip/hip_runtime.h>
#include <hip/hip_bf16.h>
#include <math.h>

// ---------------------------------------------------------------------------
// StandardTopKMoE for MI455X (gfx1250): sparse-routed top-2 MoE.
//  - bf16 WMMA (v_wmma_f32_16x16x32_bf16) with fp32 accumulate
//  - fused GEMM1 + exact GELU + GEMM2 per 32-token tile (no h round-trip)
//  - A fragments register-resident; weights pre-packed to B-fragment-tiled
//    bf16 (1 KB contiguous per 32x16 fragment) -> 1 b128-pair load per WMMA,
//    L2-resident (134 MB < 192 MB L2). Fallback direct-fp32 path if ws small.
// ---------------------------------------------------------------------------

typedef __bf16 bf16_t;
typedef bf16_t v16bf __attribute__((ext_vector_type(16)));
typedef float  v8f   __attribute__((ext_vector_type(8)));

#define B_     2
#define T_     2048
#define D_     1024
#define E_     8
#define K_     2
#define FF_    4096
#define NTOK   (B_ * T_)        // 4096 tokens
#define TILE_M 32
#define MAXTOK NTOK             // per-expert list capacity
#define TILES  (NTOK / TILE_M)  // 128

// d_out layout (flat, reference return order)
#define OFF_Y      0
#define N_Y        (NTOK * D_)
#define OFF_LOGITS (OFF_Y + N_Y)
#define N_LOGITS   (NTOK * E_)
#define OFF_IDX    (OFF_LOGITS + N_LOGITS)
#define N_IDX      (NTOK * K_)
#define OFF_CNT    (OFF_IDX + N_IDX)

// packed-weight tile counts: fragment = 32(K) x 16(N) bf16 = 512 elems = 1 KB
#define W1_FT (FF_ / 16)   // 256 N-tiles
#define W1_KT (D_ / 32)    // 32  K-tiles
#define W2_DT (D_ / 16)    // 64  N-tiles
#define W2_KT (FF_ / 32)   // 128 K-tiles
#define TILES_PER_W 65536  // E * 8192 for each of w1/w2

// A-fragment map used here for V_WMMA_F32_16X16X32_BF16 (contiguous k order, same as packed B):
// element (m, kk) -> lane = (m&15) | (kk&16), slot = kk&15
__device__ __forceinline__ void amap(int m, int kk, int& ln, int& sl) {
    ln = (m & 15) | (kk & 16);
    sl = kk & 15;
}

// ---------------------------------------------------------------------------
// Kernel 0: zero output accumulator, expert counts, routing counters
// ---------------------------------------------------------------------------
__global__ void moe_zero_kernel(float* __restrict__ out, int* __restrict__ ws_cnt) {
    size_t i = (size_t)blockIdx.x * blockDim.x + threadIdx.x;
    size_t stride = (size_t)gridDim.x * blockDim.x;
    for (size_t j = i; j < (size_t)N_Y; j += stride) out[OFF_Y + j] = 0.0f;
    if (i < E_) { out[OFF_CNT + i] = 0.0f; ws_cnt[i] = 0; }
}

// ---------------------------------------------------------------------------
// Kernel 1: gating — wave per token (coalesced float4 reads of x),
// cross-lane reduction of the 8 logits, top-2 softmax, routing lists.
// ---------------------------------------------------------------------------
__global__ __launch_bounds__(256)
void moe_gate_kernel(const float* __restrict__ x,
                     const float* __restrict__ gate_w,
                     const float* __restrict__ gate_b,
                     float* __restrict__ logits_out,
                     int*   __restrict__ idx_out,
                     float* __restrict__ cnt_out,
                     int*   __restrict__ ws_cnt,
                     int*   __restrict__ ws_tok,
                     float* __restrict__ ws_wt) {
    const int lane = threadIdx.x & 31;
    const int t = blockIdx.x * (blockDim.x >> 5) + (threadIdx.x >> 5);
    if (t >= NTOK) return;

    float acc[E_];
#pragma unroll
    for (int j = 0; j < E_; ++j) acc[j] = 0.0f;

    const float4* xr = (const float4*)(x + (size_t)t * D_);
#pragma unroll
    for (int i = lane; i < D_ / 4; i += 32) {
        float4 xv = xr[i];
        int d0 = i * 4;
        const float xs[4] = { xv.x, xv.y, xv.z, xv.w };
#pragma unroll
        for (int c = 0; c < 4; ++c)
#pragma unroll
            for (int j = 0; j < E_; ++j)
                acc[j] += xs[c] * gate_w[(d0 + c) * E_ + j];
    }
#pragma unroll
    for (int j = 0; j < E_; ++j) {
#pragma unroll
        for (int off = 16; off >= 1; off >>= 1)
            acc[j] += __shfl_xor(acc[j], off, 32);
    }

    if (lane == 0) {
#pragma unroll
        for (int j = 0; j < E_; ++j) acc[j] += gate_b[j];
#pragma unroll
        for (int j = 0; j < E_; ++j) logits_out[(size_t)t * E_ + j] = acc[j];

        int i0 = 0; float v0 = acc[0];
#pragma unroll
        for (int j = 1; j < E_; ++j) if (acc[j] > v0) { v0 = acc[j]; i0 = j; }
        int i1 = (i0 == 0) ? 1 : 0; float v1 = acc[i1];
#pragma unroll
        for (int j = 0; j < E_; ++j) if (j != i0 && acc[j] > v1) { v1 = acc[j]; i1 = j; }

        idx_out[t * K_ + 0] = i0;
        idx_out[t * K_ + 1] = i1;

        float w1x = __expf(v1 - v0);
        float inv = 1.0f / (1.0f + w1x);
        float p0 = inv, p1 = w1x * inv;

        atomicAdd(&cnt_out[i0], 1.0f);
        atomicAdd(&cnt_out[i1], 1.0f);

        int q0 = atomicAdd(&ws_cnt[i0], 1);
        ws_tok[i0 * MAXTOK + q0] = t;  ws_wt[i0 * MAXTOK + q0] = p0;
        int q1 = atomicAdd(&ws_cnt[i1], 1);
        ws_tok[i1 * MAXTOK + q1] = t;  ws_wt[i1 * MAXTOK + q1] = p1;
    }
}

// ---------------------------------------------------------------------------
// Kernel 1b: pack w1/w2 fp32 -> bf16 in B-fragment tile layout.
//   One wave per 32x16 fragment; lane l owns col n=l&15, rows kb+(l>>4)*16+s.
//   Output: tile base contiguous 1 KB, lane stride 32 B -> hot loop loads the
//   whole fragment as one aligned v16bf (2x global_load_b128).
// ---------------------------------------------------------------------------
__global__ __launch_bounds__(256)
void moe_pack_kernel(const float* __restrict__ w1,
                     const float* __restrict__ w2,
                     bf16_t* __restrict__ w1t,
                     bf16_t* __restrict__ w2t) {
    int gwave = blockIdx.x * 8 + (threadIdx.x >> 5);
    int lane  = threadIdx.x & 31;
    int nl = lane & 15, ksub = (lane >> 4) << 4;
    v16bf v;
    if (gwave < TILES_PER_W) {
        // w1 tile id = (e*W1_FT + ft)*W1_KT + kt
        int id = gwave;
        int kt = id & 31, ft = (id >> 5) & 255, e = id >> 13;
        const float* src = w1 + (size_t)e * D_ * FF_;
        int n = ft * 16 + nl, kb = kt * 32 + ksub;
#pragma unroll
        for (int s = 0; s < 16; ++s) v[s] = (bf16_t)src[(size_t)(kb + s) * FF_ + n];
        *(v16bf*)(w1t + (size_t)id * 512 + lane * 16) = v;
    } else {
        // w2 tile id = (e*W2_KT + ktf)*W2_DT + dt
        int id = gwave - TILES_PER_W;
        int dt = id & 63, ktf = (id >> 6) & 127, e = id >> 13;
        const float* src = w2 + (size_t)e * FF_ * D_;
        int n = dt * 16 + nl, kb = ktf * 32 + ksub;
#pragma unroll
        for (int s = 0; s < 16; ++s) v[s] = (bf16_t)src[(size_t)(kb + s) * D_ + n];
        *(v16bf*)(w2t + (size_t)id * 512 + lane * 16) = v;
    }
}

// ---------------------------------------------------------------------------
// Kernel 2: fused expert FFN. Block = (32-token tile, expert), 8 waves.
//   GEMM1: K split across waves, LDS cross-wave reduce + bias + exact GELU.
//   GEMM2: output columns split across waves, acc in 128 VGPRs/lane.
//   PACKED=true : B fragments from pre-tiled bf16 (1 aligned v16bf load each)
//   PACKED=false: B fragments gathered from fp32 (16 b32 + cvt each)
// ---------------------------------------------------------------------------
template <bool PACKED>
__global__ __launch_bounds__(256)
void moe_ffn_kernel(const float* __restrict__ x,
                    const float* __restrict__ w1,
                    const float* __restrict__ b1,
                    const float* __restrict__ w2,
                    const float* __restrict__ b2,
                    const bf16_t* __restrict__ w1t,
                    const bf16_t* __restrict__ w2t,
                    float* __restrict__ out_y,
                    const int*   __restrict__ ws_cnt,
                    const int*   __restrict__ ws_tok,
                    const float* __restrict__ ws_wt) {
    const int e    = blockIdx.y;
    const int tile = blockIdx.x;
    const int cnt  = ws_cnt[e];
    if (tile * TILE_M >= cnt) return;
    const int vc   = min(TILE_M, cnt - tile * TILE_M);
    const int tid  = threadIdx.x;
    const int wave = tid >> 5, lane = tid & 31;

    __shared__ float  h_part[8][4][32][8];              // 32 KB GEMM1 partials
    __shared__ alignas(32) bf16_t h_alay[2][32][16];    //  2 KB gelu(h), A layout
    __shared__ int    tok_s[TILE_M];
    __shared__ float  wt_s[TILE_M];

    if (tid < TILE_M) {
        int m = tid, src = e * MAXTOK + tile * TILE_M + m;
        tok_s[m] = (m < vc) ? ws_tok[src] : 0;
        wt_s[m]  = (m < vc) ? ws_wt[src]  : 0.0f;
    }
    __syncthreads();

    // gather this wave's A fragments (K slice [wave*128,+128)) into registers
    const int mloc  = lane & 15;
    const int kbit3 = (lane >> 4) & 1;
    v16bf afrag[2][4];
#pragma unroll
    for (int mt = 0; mt < 2; ++mt) {
        int m = mt * 16 + mloc;
        bool valid = (m < vc);
        const float* xrow = x + (size_t)tok_s[m] * D_;
#pragma unroll
        for (int ks = 0; ks < 4; ++ks) {
            int kc = (wave * 4 + ks) * 32;
#pragma unroll
            for (int s = 0; s < 16; ++s) {
                int kk = s | (kbit3 << 4);
                afrag[mt][ks][s] = valid ? (bf16_t)xrow[kc + kk] : (bf16_t)0.0f;
            }
        }
    }

    const float* w1e = w1 + (size_t)e * D_ * FF_;
    const float* w2e = w2 + (size_t)e * FF_ * D_;
    const float* b1e = b1 + (size_t)e * FF_;
    const float* b2e = b2 + (size_t)e * D_;

    v8f acc[2][8];
#pragma unroll
    for (int mt = 0; mt < 2; ++mt)
#pragma unroll
        for (int ct = 0; ct < 8; ++ct) acc[mt][ct] = {};

    const int kbl = (lane >> 4) << 4;
    const int nl  = lane & 15;

#pragma unroll 1
    for (int ff0 = 0; ff0 < FF_; ff0 += 32) {
        // ---- GEMM1 partials
        v8f p[2][2] = {};
#pragma unroll
        for (int ks = 0; ks < 4; ++ks) {
            v16bf bA, bB;
            if constexpr (PACKED) {
                int ft0 = ff0 >> 4;
                int kt  = (wave << 2) + ks;
                const bf16_t* t0 = w1t
                    + ((size_t)((e * W1_FT + ft0) * W1_KT + kt)) * 512 + lane * 16;
                bA = *(const v16bf*)t0;
                bB = *(const v16bf*)(t0 + (size_t)W1_KT * 512);  // ft0+1
                __builtin_prefetch(t0 + (size_t)2 * W1_KT * 512, 0, 1);
            } else {
                int kb = (wave * 4 + ks) * 32 + kbl;
                int n0 = ff0 + nl;
#pragma unroll
                for (int s = 0; s < 16; ++s) {
                    bA[s] = (bf16_t)w1e[(size_t)(kb + s) * FF_ + n0];
                    bB[s] = (bf16_t)w1e[(size_t)(kb + s) * FF_ + n0 + 16];
                }
                __builtin_prefetch(&w1e[(size_t)kb * FF_ + n0 + 32], 0, 1);
            }
#pragma unroll
            for (int mt = 0; mt < 2; ++mt) {   // reuse bA/bB across row-tiles
                p[mt][0] = __builtin_amdgcn_wmma_f32_16x16x32_bf16(false, afrag[mt][ks], false, bA, (short)0, p[mt][0], false, false);
                p[mt][1] = __builtin_amdgcn_wmma_f32_16x16x32_bf16(false, afrag[mt][ks], false, bB, (short)0, p[mt][1], false, false);
            }
        }
#pragma unroll
        for (int mt = 0; mt < 2; ++mt)
#pragma unroll
            for (int tN = 0; tN < 2; ++tN)
#pragma unroll
                for (int i = 0; i < 8; ++i)
                    h_part[wave][mt * 2 + tN][lane][i] = p[mt][tN][i];
        __syncthreads();

        // ---- cross-wave reduce + bias + exact GELU -> bf16 A layout
#pragma unroll
        for (int r = 0; r < 4; ++r) {
            int eidx = tid + (r << 8);
            int m = eidx >> 5, kk = eidx & 31;
            int mt = m >> 4, mm = m & 15;
            int tN = kk >> 4, nn = kk & 15;
            int ln = nn + ((mm >> 3) << 4), ii = mm & 7;
            float s = 0.0f;
#pragma unroll
            for (int w = 0; w < 8; ++w) s += h_part[w][mt * 2 + tN][ln][ii];
            s += b1e[ff0 + kk];
            float g = 0.5f * s * (1.0f + erff(s * 0.70710678118f));
            int ln2, sl2;
            amap(mm, kk, ln2, sl2);
            h_alay[mt][ln2][sl2] = (bf16_t)g;
        }
        __syncthreads();

        // ---- GEMM2: wave owns output cols [wave*128,+128)
        v16bf ha0 = *(const v16bf*)&h_alay[0][lane][0];
        v16bf ha1 = *(const v16bf*)&h_alay[1][lane][0];
#pragma unroll
        for (int ct = 0; ct < 8; ++ct) {
            v16bf bb;
            if constexpr (PACKED) {
                int ktf = ff0 >> 5;
                int dt  = (wave << 3) + ct;
                const bf16_t* t2 = w2t
                    + ((size_t)((e * W2_KT + ktf) * W2_DT + dt)) * 512 + lane * 16;
                bb = *(const v16bf*)t2;
                __builtin_prefetch(t2 + (size_t)W2_DT * 512, 0, 1);  // next ktf
            } else {
                int kb2 = ff0 + kbl;
                int nb  = (wave << 7) + nl;
#pragma unroll
                for (int s = 0; s < 16; ++s)
                    bb[s] = (bf16_t)w2e[(size_t)(kb2 + s) * D_ + nb + (ct << 4)];
            }
            acc[0][ct] = __builtin_amdgcn_wmma_f32_16x16x32_bf16(false, ha0, false, bb, (short)0, acc[0][ct], false, false);
            acc[1][ct] = __builtin_amdgcn_wmma_f32_16x16x32_bf16(false, ha1, false, bb, (short)0, acc[1][ct], false, false);
        }
    }

    // ---- epilogue: weighted scatter-accumulate into output
#pragma unroll
    for (int ct = 0; ct < 8; ++ct) {
        int d = (wave << 7) + (ct << 4) + nl;
        float bbias = b2e[d];
#pragma unroll
        for (int mt = 0; mt < 2; ++mt)
#pragma unroll
            for (int i = 0; i < 8; ++i) {
                int m = mt * 16 + i + ((lane >> 4) << 3);
                if (m < vc)
                    atomicAdd(&out_y[(size_t)tok_s[m] * D_ + d],
                              wt_s[m] * (acc[mt][ct][i] + bbias));
            }
    }
}

// ---------------------------------------------------------------------------
extern "C" void kernel_launch(void* const* d_in, const int* in_sizes, int n_in,
                              void* d_out, int out_size, void* d_ws, size_t ws_size,
                              hipStream_t stream) {
    const float* x  = (const float*)d_in[0];
    const float* gw = (const float*)d_in[1];
    const float* gb = (const float*)d_in[2];
    const float* w1 = (const float*)d_in[3];
    const float* b1 = (const float*)d_in[4];
    const float* w2 = (const float*)d_in[5];
    const float* b2 = (const float*)d_in[6];

    float* out    = (float*)d_out;
    float* logits = out + OFF_LOGITS;
    int*   idx    = (int*)(out + OFF_IDX);
    float* cntf   = out + OFF_CNT;

    int*   ws_cnt = (int*)d_ws;
    int*   ws_tok = ws_cnt + 16;
    float* ws_wt  = (float*)(ws_tok + E_ * MAXTOK);

    const size_t ROUTE_BYTES = 512 * 1024;                      // routing region
    const size_t WT_ELEMS    = (size_t)E_ * D_ * FF_;           // per weight
    const size_t NEED        = ROUTE_BYTES + 2 * WT_ELEMS * sizeof(bf16_t); // ~134.7 MB
    bf16_t* w1t = (bf16_t*)((char*)d_ws + ROUTE_BYTES);
    bf16_t* w2t = w1t + WT_ELEMS;
    const bool packed = (ws_size >= NEED);

    moe_zero_kernel<<<1024, 256, 0, stream>>>(out, ws_cnt);
    moe_gate_kernel<<<NTOK / 8, 256, 0, stream>>>(x, gw, gb, logits, idx, cntf,
                                                  ws_cnt, ws_tok, ws_wt);
    dim3 grid(TILES, E_);
    if (packed) {
        moe_pack_kernel<<<(2 * TILES_PER_W) / 8, 256, 0, stream>>>(w1, w2, w1t, w2t);
        moe_ffn_kernel<true><<<grid, 256, 0, stream>>>(x, w1, b1, w2, b2, w1t, w2t,
                                                       out, ws_cnt, ws_tok, ws_wt);
    } else {
        moe_ffn_kernel<false><<<grid, 256, 0, stream>>>(x, w1, b1, w2, b2, w1t, w2t,
                                                        out, ws_cnt, ws_tok, ws_wt);
    }
}
